// ProbEncoderTDFF_59219009077528
// MI455X (gfx1250) — compile-verified, matrix operand order checked by reading, not run on hardware
//
#include <hip/hip_runtime.h>

#define LL   512
#define BB   2
#define SS   432
#define DM   64
#define RK   256
#define NROW (BB*LL)   // 1024

typedef __attribute__((ext_vector_type(16))) __bf16         v16bf;
typedef __attribute__((ext_vector_type(8)))  __bf16         v8bf;
typedef __attribute__((ext_vector_type(16))) unsigned short v16u;
typedef __attribute__((ext_vector_type(8)))  float          v8f;

__device__ __forceinline__ v8f vz8() {
  v8f z = {0.f, 0.f, 0.f, 0.f, 0.f, 0.f, 0.f, 0.f};
  return z;
}

__device__ __forceinline__ v8f wmma_bf(v16bf a, v16bf b, v8f c) {
  return __builtin_amdgcn_wmma_f32_16x16x32_bf16(false, a, false, b, (short)0, c, false, false);
}

// ---- CDNA5 async global->LDS (tracked by ASYNCcnt) ----
__device__ __forceinline__ void async_b128(unsigned lds_addr, const void* gptr) {
  unsigned long long ga = (unsigned long long)reinterpret_cast<unsigned long long>(gptr);
  asm volatile("global_load_async_to_lds_b128 %0, %1, off" : : "v"(lds_addr), "v"(ga) : "memory");
}
__device__ __forceinline__ void wait_async0() {
  asm volatile("s_wait_asynccnt 0x0" ::: "memory");
}

// ---- Fragment builders (wave32, 16x16x32 bf16 WMMA) ----
// Element e of a 16-half A/B fragment maps to K = k0 + 8h + e (e<8)
// and K = k0 + 16 + 8h + (e-8) (e>=8), with h = lane>>4 (ISA 7.12.2).

// A fragment = elementwise product of two bf16 rows (native pk bf16 math,
// results are already packed fragment halves - no f32 round trip).
__device__ __forceinline__ v16bf afrag_mulbf(const __bf16* row, const __bf16* scale, int k0, int h) {
  const int b0 = k0 + (h << 3);
  const int b1 = b0 + 16;
  v8bf x0 = *(const v8bf*)(row + b0);
  v8bf x1 = *(const v8bf*)(row + b1);
  v8bf s0 = *(const v8bf*)(scale + b0);
  v8bf s1 = *(const v8bf*)(scale + b1);
  v8bf p0 = x0 * s0;
  v8bf p1 = x1 * s1;
  return __builtin_shufflevector(p0, p1, 0, 1, 2, 3, 4, 5, 6, 7, 8, 9, 10, 11, 12, 13, 14, 15);
}

// A fragment from a bf16 row (no scaling)
__device__ __forceinline__ v16bf afrag_bf(const __bf16* row, int k0, int h) {
  const int b0 = k0 + (h << 3);
  const int b1 = b0 + 16;
  v8bf x0 = *(const v8bf*)(row + b0);
  v8bf x1 = *(const v8bf*)(row + b1);
  return __builtin_shufflevector(x0, x1, 0, 1, 2, 3, 4, 5, 6, 7, 8, 9, 10, 11, 12, 13, 14, 15);
}

// A fragment from an f32 row (native f32->bf16 converts)
__device__ __forceinline__ v16bf afrag_f32(const float* row, int k0, int h) {
  const int b0 = k0 + (h << 3);
  const int b1 = b0 + 16;
  v16bf u;
#pragma unroll
  for (int e = 0; e < 8; e++) {
    u[e]     = (__bf16)row[b0 + e];
    u[8 + e] = (__bf16)row[b1 + e];
  }
  return u;
}

// B fragment where B[k][n] = M[(n)*ld + k]  (K contiguous within a row of M)
__device__ __forceinline__ v16bf bfrag_rowK(const unsigned short* M, int ld, int nrow, int k0, int h) {
  const __bf16* p = (const __bf16*)(M + nrow * ld + k0 + (h << 3));
  v8bf x0 = *(const v8bf*)(p);
  v8bf x1 = *(const v8bf*)(p + 16);
  return __builtin_shufflevector(x0, x1, 0, 1, 2, 3, 4, 5, 6, 7, 8, 9, 10, 11, 12, 13, 14, 15);
}

// B fragment where B[k][n] = M[(k)*ld + n]  (K is the row dimension of M)
__device__ __forceinline__ v16bf bfrag_colK(const unsigned short* M, int ld, int k0, int ncol, int h) {
  const int b0 = k0 + (h << 3);
  const int b1 = b0 + 16;
  v16bf u;
#pragma unroll
  for (int e = 0; e < 8; e++) {
    u[e]     = ((const __bf16*)M)[(b0 + e) * ld + ncol];
    u[8 + e] = ((const __bf16*)M)[(b1 + e) * ld + ncol];
  }
  return u;
}

// ---------------------------------------------------------------------------
// Prep: bf16 copies of U0,U1,V0,V1,W0,W1 (64x256), W0^T,W1^T (256x64), col-sums of W
__global__ __launch_bounds__(256) void prep_kernel(
    const float* U, const float* V, const float* W,
    unsigned short* U0b, unsigned short* U1b,
    unsigned short* V0b, unsigned short* V1b,
    unsigned short* W0b, unsigned short* W1b,
    unsigned short* W0Tb, unsigned short* W1Tb,
    float* w0bar, float* w1bar) {
  int idx = blockIdx.x * 256 + threadIdx.x;       // 64 blocks -> 16384
  if (idx < DM * RK) {
    int c = idx >> 8, d = idx & 255;
    ((__bf16*)U0b)[idx] = (__bf16)U[idx];
    ((__bf16*)U1b)[idx] = (__bf16)U[DM * RK + idx];
    ((__bf16*)V0b)[idx] = (__bf16)V[idx];
    ((__bf16*)V1b)[idx] = (__bf16)V[DM * RK + idx];
    float w0 = W[idx], w1 = W[DM * RK + idx];
    ((__bf16*)W0b)[idx] = (__bf16)w0;
    ((__bf16*)W1b)[idx] = (__bf16)w1;
    ((__bf16*)W0Tb)[d * DM + c] = (__bf16)w0;
    ((__bf16*)W1Tb)[d * DM + c] = (__bf16)w1;
  }
  if (blockIdx.x == 0) {
    int d = threadIdx.x;                           // 256 columns
    float s0 = 0.f, s1 = 0.f;
#pragma unroll 4
    for (int c = 0; c < DM; c++) {
      s0 += W[c * RK + d];
      s1 += W[DM * RK + c * RK + d];
    }
    w0bar[d] = s0;
    w1bar[d] = s1;
  }
}

// xc = concat(x, ff broadcast), qz = xc
__global__ __launch_bounds__(64) void init_kernel(const float* x, const float* ff, float* xc, float* qz) {
  int row = blockIdx.x;                            // 0..1023
  int e = threadIdx.x;                             // 0..63
  int z = row >> 9, i = row & (LL - 1);
  float v = (i < SS) ? x[(z * SS + i) * DM + e] : ff[(i - SS) * DM + e];
  xc[row * DM + e] = v;
  qz[row * DM + e] = v;
}

// row softmax of q_z, times mask, emitted as bf16 (WMMA A operand)
__global__ __launch_bounds__(32) void softmax_kernel(const float* qz, const int* mask, unsigned short* Zb) {
  int row = blockIdx.x;                            // 0..1023
  int lane = threadIdx.x;                          // 0..31 (wave32)
  float a = qz[row * DM + lane];
  float b = qz[row * DM + 32 + lane];
  float m = fmaxf(a, b);
#pragma unroll
  for (int off = 16; off; off >>= 1) m = fmaxf(m, __shfl_xor(m, off, 32));
  float ea = __expf(a - m), eb = __expf(b - m);
  float s = ea + eb;
#pragma unroll
  for (int off = 16; off; off >>= 1) s += __shfl_xor(s, off, 32);
  float inv = (mask[row] != 0 ? 1.f : 0.f) / s;
  ((__bf16*)Zb)[row * DM + lane]      = (__bf16)(ea * inv);
  ((__bf16*)Zb)[row * DM + 32 + lane] = (__bf16)(eb * inv);
}

// out[1024x256] = Zb[1024x64] @ Mb[64x256]; emits f32 and bf16 copies
__global__ __launch_bounds__(32) void gemm_vec_kernel(const unsigned short* Zb, const unsigned short* Mb,
                                                      float* out, unsigned short* outb) {
  int lane = threadIdx.x & 31;
  int r = lane & 15, h = (lane >> 4) & 1;
  int mt = blockIdx.x;                             // 0..63
  int nt = blockIdx.y;                             // 0..15
  const __bf16* arow = (const __bf16*)Zb + (mt * 16 + r) * DM;
  v8f acc = vz8();
#pragma unroll
  for (int kt = 0; kt < 2; kt++) {
    v16bf a = afrag_bf(arow, kt * 32, h);
    v16bf b = bfrag_colK(Mb, RK, kt * 32, nt * 16 + r, h);
    acc = wmma_bf(a, b, acc);
  }
#pragma unroll
  for (int v = 0; v < 8; v++) {
    int ofs = (mt * 16 + v + 8 * h) * RK + nt * 16 + r;
    out[ofs] = acc[v];
    ((__bf16*)outb)[ofs] = (__bf16)acc[v];
  }
}

// iteration 0: softmax(ones)=1/64 -> msg = (1/64)*wbar .* (masked colsum - own)
__global__ __launch_bounds__(256) void colsum_kernel(const float* zv0, const float* zv1, const int* mask,
                                                     float* cs0, float* cs1) {
  int z = blockIdx.x;                              // 0..1
  int d = threadIdx.x;                             // 0..255
  float s0 = 0.f, s1 = 0.f;
  for (int j = 0; j < LL; j++) {
    float mk = (mask[z * LL + j] != 0) ? 1.f : 0.f;
    s0 += mk * zv0[(z * LL + j) * RK + d];
    s1 += mk * zv1[(z * LL + j) * RK + d];
  }
  cs0[z * RK + d] = s0;
  cs1[z * RK + d] = s1;
}

__global__ __launch_bounds__(256) void msg_uniform_kernel(
    const float* zv0, const float* zv1, const float* cs0, const float* cs1,
    const float* w0bar, const float* w1bar, const int* mask, float* msg0, float* msg1) {
  int row = blockIdx.x;                            // 0..1023
  int d = threadIdx.x;                             // 0..255
  int z = row >> 9;
  float mq = (mask[row] != 0) ? 1.f : 0.f;
  const float inv64 = 1.f / 64.f;
  msg0[row * RK + d] = mq * inv64 * w0bar[d] * (cs0[z * RK + d] - mq * zv0[row * RK + d]);
  msg1[row * RK + d] = mq * inv64 * w1bar[d] * (cs1[z * RK + d] - mq * zv1[row * RK + d]);
}

// ---------------------------------------------------------------------------
// Fused message kernel (the hot loop).  For each output row q:
//   P[p,c]  = sum_d X[p,d]*x[q,d]*W0[c,d] + sum_d Y[p,d]*y[q,d]*W1[c,d]
//   s[p,c]  = softmax_c(P) * mask(p) * mask(q) * [p != q]
//   msg[q,d] += sum_p ( s[p,:] @ Wsel )[d] * ZV[p,d]
// One wave per q; 4 waves/WG share double-buffered LDS tiles filled with
// CDNA5 async global->LDS DMA, software-pipelined against the WMMA work.
__global__ __launch_bounds__(128) void msg_fused_kernel(
    const unsigned short* __restrict__ Xpb, const unsigned short* __restrict__ xqb_all,
    const unsigned short* __restrict__ Ypb, const unsigned short* __restrict__ yqb_all,
    const unsigned short* __restrict__ W0b, const unsigned short* __restrict__ W1b,
    const unsigned short* __restrict__ WselTb,     // [256 x 64] = Wsel^T, bf16
    const float* __restrict__ ZV, const int* __restrict__ mask,
    float* __restrict__ msgOut) {
  __shared__ __bf16 tXb[2][16 * RK];               // 2 x 8 KB
  __shared__ __bf16 tYb[2][16 * RK];               // 2 x 8 KB
  __shared__ float  tZ [2][16 * RK];               // 2 x 16 KB
  __shared__ __bf16 sS[4][16 * DM];                // 8 KB

  const int tid  = threadIdx.x;
  const int w    = tid >> 5;
  const int lane = tid & 31;
  const int r    = lane & 15;
  const int h    = (lane >> 4) & 1;
  const int q    = blockIdx.x * 4 + w;             // 0..1023 (same z within WG)
  const int z    = q >> 9;
  const int qloc = q & (LL - 1);
  const __bf16* xq = (const __bf16*)xqb_all + q * RK;
  const __bf16* yq = (const __bf16*)yqb_all + q * RK;
  const float   mq = (mask[q] != 0) ? 1.f : 0.f;
  __bf16* sw = &sS[w][0];

  const unsigned ldsX = (unsigned)reinterpret_cast<unsigned long long>(&tXb[0][0]);
  const unsigned ldsY = (unsigned)reinterpret_cast<unsigned long long>(&tYb[0][0]);
  const unsigned ldsZ = (unsigned)reinterpret_cast<unsigned long long>(&tZ[0][0]);

  float msgReg[16];
#pragma unroll
  for (int nt = 0; nt < 16; nt++) msgReg[nt] = 0.f;

  // async prefetch of one 16-row p-tile (split across the 128 threads)
  auto prefetch = [&](int pt, int buf) {
    const char* gX = (const char*)(Xpb + (size_t)(z * LL + pt * 16) * RK);
    const char* gY = (const char*)(Ypb + (size_t)(z * LL + pt * 16) * RK);
    const char* gZ = (const char*)(ZV  + (size_t)(z * LL + pt * 16) * RK);
    unsigned bX = ldsX + (unsigned)buf * (16 * RK * 2);
    unsigned bY = ldsY + (unsigned)buf * (16 * RK * 2);
    unsigned bZ = ldsZ + (unsigned)buf * (16 * RK * 4);
#pragma unroll
    for (int i = 0; i < 4; i++) {                  // 512 x b128 for X and Y (bf16)
      int e = tid + i * 128;
      async_b128(bX + e * 16, gX + e * 16);
      async_b128(bY + e * 16, gY + e * 16);
    }
#pragma unroll
    for (int i = 0; i < 8; i++) {                  // 1024 x b128 for Z (f32)
      int e = tid + i * 128;
      async_b128(bZ + e * 16, gZ + e * 16);
    }
  };

  prefetch(0, 0);

  for (int pt = 0; pt < LL / 16; pt++) {
    const int buf = pt & 1;
    wait_async0();                                 // my part of tile pt landed
    __syncthreads();                               // all parts landed; buf^1 free
    if (pt + 1 < LL / 16) prefetch(pt + 1, buf ^ 1); // overlap DMA with compute

    // ---- P tile [16p x 64c] in 4 f32 accumulators
    v8f acc[4];
#pragma unroll
    for (int ct = 0; ct < 4; ct++) acc[ct] = vz8();
    const __bf16* Xrow = &tXb[buf][r * RK];
    const __bf16* Yrow = &tYb[buf][r * RK];
#pragma unroll
    for (int kt = 0; kt < 8; kt++) {
      v16bf aX = afrag_mulbf(Xrow, xq, kt * 32, h);
      v16bf aY = afrag_mulbf(Yrow, yq, kt * 32, h);
#pragma unroll
      for (int ct = 0; ct < 4; ct++) {
        acc[ct] = wmma_bf(aX, bfrag_rowK(W0b, RK, ct * 16 + r, kt * 32, h), acc[ct]);
        acc[ct] = wmma_bf(aY, bfrag_rowK(W1b, RK, ct * 16 + r, kt * 32, h), acc[ct]);
      }
    }

    // ---- softmax over c (per p row), apply off-diagonal + mask, stash as bf16
#pragma unroll
    for (int v = 0; v < 8; v++) {
      float m = fmaxf(fmaxf(acc[0][v], acc[1][v]), fmaxf(acc[2][v], acc[3][v]));
#pragma unroll
      for (int off = 8; off; off >>= 1) m = fmaxf(m, __shfl_xor(m, off, 32));
      float e0 = __expf(acc[0][v] - m), e1 = __expf(acc[1][v] - m);
      float e2 = __expf(acc[2][v] - m), e3 = __expf(acc[3][v] - m);
      float s = e0 + e1 + e2 + e3;
#pragma unroll
      for (int off = 8; off; off >>= 1) s += __shfl_xor(s, off, 32);
      int pl = pt * 16 + v + 8 * h;                // local p index
      float mp = (mask[z * LL + pl] != 0) ? 1.f : 0.f;
      float f = (pl == qloc) ? 0.f : (mq * mp) / s;
      int ro = (v + 8 * h) * DM + r;
      sw[ro +  0] = (__bf16)(e0 * f);
      sw[ro + 16] = (__bf16)(e1 * f);
      sw[ro + 32] = (__bf16)(e2 * f);
      sw[ro + 48] = (__bf16)(e3 * f);
    }
    asm volatile("s_wait_dscnt 0x0" ::: "memory"); // per-wave LDS store->load

    // ---- sc = s @ Wsel  (16p x 256d), folded into msgReg with ZV weight
    v16bf sa0 = afrag_bf(sw + r * DM, 0, h);
    v16bf sa1 = afrag_bf(sw + r * DM, 32, h);
    const float* Zt = &tZ[buf][0];
#pragma unroll
    for (int nt = 0; nt < 16; nt++) {
      v8f a2 = vz8();
      a2 = wmma_bf(sa0, bfrag_rowK(WselTb, DM, nt * 16 + r, 0, h), a2);
      a2 = wmma_bf(sa1, bfrag_rowK(WselTb, DM, nt * 16 + r, 32, h), a2);
      int dcol = nt * 16 + r;
      float am = 0.f;
#pragma unroll
      for (int v = 0; v < 8; v++) am += a2[v] * Zt[(v + 8 * h) * RK + dcol];
      msgReg[nt] += am;
    }
    __syncthreads();                               // all waves done with buf
  }

#pragma unroll
  for (int nt = 0; nt < 16; nt++) msgReg[nt] += __shfl_xor(msgReg[nt], 16, 32);
  if (lane < 16) {
#pragma unroll
    for (int nt = 0; nt < 16; nt++)
      msgOut[q * RK + nt * 16 + lane] = msgReg[nt];
  }
}

// q_z = xc + msg0 @ U0^T + msg1 @ U1^T
__global__ __launch_bounds__(32) void update_kernel(
    const float* msg0, const float* msg1,
    const unsigned short* U0b, const unsigned short* U1b,
    const float* xc, float* qz) {
  int lane = threadIdx.x & 31;
  int r = lane & 15, h = (lane >> 4) & 1;
  int mt = blockIdx.x;                             // 0..63
  int nt = blockIdx.y;                             // 0..3
  const float* m0row = msg0 + (mt * 16 + r) * RK;
  const float* m1row = msg1 + (mt * 16 + r) * RK;
  v8f acc = vz8();
#pragma unroll
  for (int kt = 0; kt < 8; kt++) {
    acc = wmma_bf(afrag_f32(m0row, kt * 32, h), bfrag_rowK(U0b, RK, nt * 16 + r, kt * 32, h), acc);
    acc = wmma_bf(afrag_f32(m1row, kt * 32, h), bfrag_rowK(U1b, RK, nt * 16 + r, kt * 32, h), acc);
  }
#pragma unroll
  for (int v = 0; v < 8; v++) {
    int row = mt * 16 + v + 8 * h;
    int col = nt * 16 + r;
    qz[row * DM + col] = xc[row * DM + col] + acc[v];
  }
}

__global__ __launch_bounds__(64) void final_kernel(const float* qz, float* out) {
  int row = blockIdx.x;                            // 0..863 (2*432)
  int z = row / SS, i = row - z * SS;
  out[row * DM + threadIdx.x] = qz[(z * LL + i) * DM + threadIdx.x];
}

// ---------------------------------------------------------------------------
extern "C" void kernel_launch(void* const* d_in, const int* in_sizes, int n_in,
                              void* d_out, int out_size, void* d_ws, size_t ws_size,
                              hipStream_t stream) {
  (void)in_sizes; (void)n_in; (void)out_size; (void)ws_size;
  const float* x    = (const float*)d_in[0];   // [2,432,64]
  const int*   mask = (const int*)d_in[1];     // [2,512]
  const float* U    = (const float*)d_in[2];   // [2,64,256]
  const float* V    = (const float*)d_in[3];   // [2,64,256]
  const float* W    = (const float*)d_in[4];   // [2,64,256]
  const float* ff   = (const float*)d_in[5];   // [80,64]
  float* out = (float*)d_out;

  char* p = (char*)d_ws;
  auto alloc = [&](size_t bytes) -> void* {
    void* r = (void*)p;
    p += (bytes + 255) & ~(size_t)255;
    return r;
  };
  float* xc  = (float*)alloc((size_t)NROW * DM * 4);
  float* qz  = (float*)alloc((size_t)NROW * DM * 4);
  unsigned short* Zb = (unsigned short*)alloc((size_t)NROW * DM * 2);
  float* vec[2][4];
  unsigned short* vecb[2][4];
  for (int s = 0; s < 2; s++)
    for (int m = 0; m < 4; m++) {
      vec[s][m]  = (float*)alloc((size_t)NROW * RK * 4);
      vecb[s][m] = (unsigned short*)alloc((size_t)NROW * RK * 2);
    }
  float* msg0 = (float*)alloc((size_t)NROW * RK * 4);
  float* msg1 = (float*)alloc((size_t)NROW * RK * 4);
  unsigned short* U0b  = (unsigned short*)alloc((size_t)DM * RK * 2);
  unsigned short* U1b  = (unsigned short*)alloc((size_t)DM * RK * 2);
  unsigned short* V0b  = (unsigned short*)alloc((size_t)DM * RK * 2);
  unsigned short* V1b  = (unsigned short*)alloc((size_t)DM * RK * 2);
  unsigned short* W0b  = (unsigned short*)alloc((size_t)DM * RK * 2);
  unsigned short* W1b  = (unsigned short*)alloc((size_t)DM * RK * 2);
  unsigned short* W0Tb = (unsigned short*)alloc((size_t)RK * DM * 2);
  unsigned short* W1Tb = (unsigned short*)alloc((size_t)RK * DM * 2);
  float* w0bar = (float*)alloc(RK * 4);
  float* w1bar = (float*)alloc(RK * 4);
  float* cs0   = (float*)alloc((size_t)BB * RK * 4);
  float* cs1   = (float*)alloc((size_t)BB * RK * 4);

  prep_kernel<<<64, 256, 0, stream>>>(U, V, W, U0b, U1b, V0b, V1b, W0b, W1b, W0Tb, W1Tb, w0bar, w1bar);
  init_kernel<<<NROW, 64, 0, stream>>>(x, ff, xc, qz);

  int cur = 0;
  for (int it = 0; it < 4; it++) {
    softmax_kernel<<<NROW, 32, 0, stream>>>(qz, mask, Zb);
    gemm_vec_kernel<<<dim3(NROW / 16, RK / 16), 32, 0, stream>>>(Zb, U0b, vec[cur][0], vecb[cur][0]); // a0
    gemm_vec_kernel<<<dim3(NROW / 16, RK / 16), 32, 0, stream>>>(Zb, V0b, vec[cur][1], vecb[cur][1]); // zv0
    gemm_vec_kernel<<<dim3(NROW / 16, RK / 16), 32, 0, stream>>>(Zb, U1b, vec[cur][2], vecb[cur][2]); // a1
    gemm_vec_kernel<<<dim3(NROW / 16, RK / 16), 32, 0, stream>>>(Zb, V1b, vec[cur][3], vecb[cur][3]); // zv1

    if (it == 0) {
      colsum_kernel<<<BB, 256, 0, stream>>>(vec[cur][1], vec[cur][3], mask, cs0, cs1);
      msg_uniform_kernel<<<NROW, 256, 0, stream>>>(vec[cur][1], vec[cur][3], cs0, cs1,
                                                   w0bar, w1bar, mask, msg0, msg1);
    } else {
      int pv = 1 - cur;
      // msg0[i,d] = sum_{j,c} s[i,j,c] W0[c,d] zv0[j,d]
      msg_fused_kernel<<<NROW / 4, 128, 0, stream>>>(
          vecb[pv][1], vecb[pv][0], vecb[pv][2], vecb[pv][3],  // X=zv0p x=a0p Y=a1p y=zv1p
          W0b, W1b, W0Tb, vec[cur][1], mask, msg0);
      // msg1[i,d] = sum_{j,c} s[j,i,c] W1[c,d] zv1[j,d]
      msg_fused_kernel<<<NROW / 4, 128, 0, stream>>>(
          vecb[pv][0], vecb[pv][1], vecb[pv][3], vecb[pv][2],  // X=a0p x=zv0p Y=zv1p y=a1p
          W0b, W1b, W1Tb, vec[cur][3], mask, msg1);
    }
    update_kernel<<<dim3(NROW / 16, DM / 16), 32, 0, stream>>>(msg0, msg1, U0b, U1b, xc, qz);
    cur ^= 1;
  }

  final_kernel<<<BB * SS, 64, 0, stream>>>(qz, out);
}
